// UniAttention_68186900791676
// MI455X (gfx1250) — matrix operand order repaired and checked by reading, not compiled, not run on hardware
//
#include <hip/hip_runtime.h>

#define HIDDEN 2048
#define NHEADS 16
#define HEADD  128
#define BBATCH 2
#define SSEQ   2048
#define MROWS  (BBATCH * SSEQ)   // 4096
#define NQKV   (3 * HIDDEN)      // 6144

typedef __bf16 v8bf  __attribute__((ext_vector_type(8)));
typedef __bf16 v16bf __attribute__((ext_vector_type(16)));
typedef float  v8f   __attribute__((ext_vector_type(8)));

#define WMMA_BF16(a, b, c) \
  __builtin_amdgcn_wmma_f32_16x16x32_bf16(false, (a), false, (b), (short)0, (c), false, false)

// ---------------------------------------------------------------------------
// WMMA fragment loaders (wave32; contiguous k order used for both A and B, not the ISA table order)
// A-matrix 16x32 bf16: lane L holds row L%16; K = (L/16)*16 + 0..15
// -> two contiguous 8-element (16B) runs.
__device__ inline v16bf load_a_frag(const __bf16* base, int row_stride, int lane) {
  const __bf16* r = base + (size_t)(lane & 15) * row_stride + ((lane >> 4) << 4);
  v8bf lo = *(const v8bf*)(r);
  v8bf hi = *(const v8bf*)(r + 8);
  return __builtin_shufflevector(lo, hi, 0, 1, 2, 3, 4, 5, 6, 7,
                                          8, 9, 10, 11, 12, 13, 14, 15);
}
// B-matrix 32x16 bf16 (K x N), sourced from transposed storage Bt[n][k]:
// lane L holds column L%16; K-run = (L/16)*16 + 0..15 -> one 32B run.
__device__ inline v16bf load_b_frag(const __bf16* baseT, int col_stride, int lane) {
  const __bf16* r = baseT + (size_t)(lane & 15) * col_stride + ((lane >> 4) << 4);
  return *(const v16bf*)r;
}

// ---------------------------------------------------------------------------
// fp32 -> bf16 elementwise convert (coalesced)
__global__ __launch_bounds__(256) void cvt_f32_to_bf16_kernel(const float* __restrict__ in,
                                                              __bf16* __restrict__ out,
                                                              long long n) {
  long long i = (long long)blockIdx.x * 256 + threadIdx.x;
  if (i < n) out[i] = (__bf16)in[i];
}

// LDS-tiled transpose: in [K][N] f32 row-major -> out [N][K] bf16.
// 32x32 tiles, 256 threads (32x8), padded LDS to avoid bank conflicts.
__global__ __launch_bounds__(256) void transpose_to_bf16_kernel(const float* __restrict__ in,
                                                                __bf16* __restrict__ out,
                                                                int K, int N) {
  __shared__ float tile[32][33];
  const int tx = threadIdx.x & 31;
  const int ty = threadIdx.x >> 5;          // 0..7
  const int ntiles_n = N >> 5;
  const int k0 = (blockIdx.x / ntiles_n) << 5;
  const int n0 = (blockIdx.x % ntiles_n) << 5;
#pragma unroll
  for (int i = 0; i < 4; ++i) {
    const int kr = ty + i * 8;
    tile[kr][tx] = in[(size_t)(k0 + kr) * N + (n0 + tx)];
  }
  __syncthreads();
#pragma unroll
  for (int i = 0; i < 4; ++i) {
    const int nr = ty + i * 8;
    out[(size_t)(n0 + nr) * K + (k0 + tx)] = (__bf16)tile[tx][nr];
  }
}

// ---------------------------------------------------------------------------
// QKV GEMM: C[4096,6144] = Xbf @ WqkvT' + b_qkv, scattered into per-head
// Q [bh][s][d], K [bh][s][d], V-transposed [bh][d][s] (all bf16).
// One wave computes a 32x64 tile (2 A-frags x 4 B-frags, 8 accumulators).
__global__ __launch_bounds__(256) void qkv_gemm_kernel(const __bf16* __restrict__ X,
                                                       const __bf16* __restrict__ Wt,
                                                       const float* __restrict__ bias,
                                                       __bf16* __restrict__ q_ws,
                                                       __bf16* __restrict__ k_ws,
                                                       __bf16* __restrict__ vt_ws) {
  const int lane = threadIdx.x & 31;
  const int wave = threadIdx.x >> 5;
  const int widx = blockIdx.x * 8 + wave;      // 12288 waves total
  const int nstrips = NQKV / 64;               // 96
  const int mt = widx / nstrips;               // 0..127
  const int ns = widx % nstrips;
  const int m0 = mt * 32, n0 = ns * 64;

  v8f acc0[4] = {};
  v8f acc1[4] = {};
  for (int kb = 0; kb < HIDDEN; kb += 32) {
    __builtin_prefetch((const void*)(X + (size_t)m0 * HIDDEN + kb + 512), 0, 0);
    v16bf a0 = load_a_frag(X + (size_t)m0 * HIDDEN + kb, HIDDEN, lane);
    v16bf a1 = load_a_frag(X + (size_t)(m0 + 16) * HIDDEN + kb, HIDDEN, lane);
#pragma unroll
    for (int j = 0; j < 4; ++j) {
      v16bf b = load_b_frag(Wt + (size_t)(n0 + j * 16) * HIDDEN + kb, HIDDEN, lane);
      acc0[j] = WMMA_BF16(a0, b, acc0[j]);
      acc1[j] = WMMA_BF16(a1, b, acc1[j]);
    }
  }

  const int lh = lane >> 4;      // which 8-row half of the C tile
  const int lc = lane & 15;      // column within tile
#pragma unroll
  for (int j = 0; j < 4; ++j) {
    const int n = n0 + j * 16 + lc;
    const float bv = bias[n];
    const int sec = n >> 11;     // 0=Q 1=K 2=V
    const int nn = n & 2047;
    const int h = nn >> 7, d = nn & 127;
#pragma unroll
    for (int mi = 0; mi < 2; ++mi) {
#pragma unroll
      for (int r = 0; r < 8; ++r) {
        const int m = m0 + mi * 16 + r + lh * 8;
        const int bb = m >> 11, s = m & 2047;
        const int bh = bb * NHEADS + h;
        const float av = (mi == 0) ? acc0[j][r] : acc1[j][r];
        const __bf16 v = (__bf16)(av + bv);
        if (sec == 0)      q_ws[((size_t)bh * SSEQ + s) * HEADD + d] = v;
        else if (sec == 1) k_ws[((size_t)bh * SSEQ + s) * HEADD + d] = v;
        else               vt_ws[((size_t)bh * HEADD + d) * SSEQ + s] = v;
      }
    }
  }
}

// ---------------------------------------------------------------------------
// Flash attention: one wave per (b,h, 16-query tile). Online softmax with
// causal mask (-10000 on scaled scores, matching the reference). Both 16-key
// score sub-tiles of a 32-key block are reduced in one fused softmax pass.
__global__ __launch_bounds__(128) void attn_kernel(const __bf16* __restrict__ q_ws,
                                                   const __bf16* __restrict__ k_ws,
                                                   const __bf16* __restrict__ vt_ws,
                                                   __bf16* __restrict__ attn_ws) {
  __shared__ __align__(32) __bf16 pbuf[4][16 * 32];
  const int lane = threadIdx.x & 31;
  const int wave = threadIdx.x >> 5;
  const int widx = blockIdx.x * 4 + wave;      // 4096 waves total
  const int qt = widx & 127;
  const int bh = widx >> 7;                    // 0..31
  const int qbase = qt * 16;
  const int qend = qbase + 15;

  const __bf16* Q  = q_ws  + (size_t)bh * SSEQ * HEADD;
  const __bf16* Km = k_ws  + (size_t)bh * SSEQ * HEADD;
  const __bf16* Vt = vt_ws + (size_t)bh * HEADD * SSEQ;
  __bf16* pb = pbuf[wave];

  const int lh = lane >> 4;
  const int lc = lane & 15;
  const float scale = 0.08838834764831845f;    // 1/sqrt(128)

  v16bf qf[4];
#pragma unroll
  for (int c = 0; c < 4; ++c)
    qf[c] = load_a_frag(Q + (size_t)qbase * HEADD + 32 * c, HEADD, lane);

  float mrow[8], lrow[8];
  v8f o[8] = {};
#pragma unroll
  for (int r = 0; r < 8; ++r) { mrow[r] = -1.0e30f; lrow[r] = 0.0f; }

  for (int kb = 0; kb <= qend; kb += 32) {
    // scores for the whole 32-key block (two 16-col WMMA tiles)
    v8f sc0 = {}, sc1 = {};
#pragma unroll
    for (int c = 0; c < 4; ++c) {
      v16bf b0 = load_b_frag(Km + (size_t)kb * HEADD + 32 * c, HEADD, lane);
      v16bf b1 = load_b_frag(Km + (size_t)(kb + 16) * HEADD + 32 * c, HEADD, lane);
      sc0 = WMMA_BF16(qf[c], b0, sc0);
      sc1 = WMMA_BF16(qf[c], b1, sc1);
    }
    const int key0 = kb + lc;
    const int key1 = kb + 16 + lc;
#pragma unroll
    for (int r = 0; r < 8; ++r) {
      const int q = qbase + r + lh * 8;
      float sv0 = sc0[r] * scale;
      float sv1 = sc1[r] * scale;
      if (q < key0) sv0 = -10000.0f;           // causal mask, like reference
      if (q < key1) sv1 = -10000.0f;
      // fused 32-wide row-max (xor 1,2,4,8 stays within the 16-lane half)
      float rm = fmaxf(sv0, sv1);
      rm = fmaxf(rm, __shfl_xor(rm, 1, 32));
      rm = fmaxf(rm, __shfl_xor(rm, 2, 32));
      rm = fmaxf(rm, __shfl_xor(rm, 4, 32));
      rm = fmaxf(rm, __shfl_xor(rm, 8, 32));
      const float mnew = fmaxf(mrow[r], rm);
      const float corr = __expf(mrow[r] - mnew);
      const float p0 = __expf(sv0 - mnew);
      const float p1 = __expf(sv1 - mnew);
      float rs = p0 + p1;
      rs += __shfl_xor(rs, 1, 32);
      rs += __shfl_xor(rs, 2, 32);
      rs += __shfl_xor(rs, 4, 32);
      rs += __shfl_xor(rs, 8, 32);
      lrow[r] = lrow[r] * corr + rs;
      mrow[r] = mnew;
#pragma unroll
      for (int j = 0; j < 8; ++j) o[j][r] *= corr;
      pb[(r + lh * 8) * 32 + lc]      = (__bf16)p0;   // C-layout -> LDS
      pb[(r + lh * 8) * 32 + 16 + lc] = (__bf16)p1;
    }
    asm volatile("s_wait_dscnt 0" ::: "memory");
    v16bf pf = load_a_frag(pb, 32, lane);      // LDS -> A-fragment (P, 16x32)
#pragma unroll
    for (int j = 0; j < 8; ++j) {
      v16bf vf = load_b_frag(Vt + (size_t)(j * 16) * SSEQ + kb, SSEQ, lane);
      o[j] = WMMA_BF16(pf, vf, o[j]);
    }
  }

  const int bb = bh >> 4, h = bh & 15;
#pragma unroll
  for (int r = 0; r < 8; ++r) {
    const float inv = 1.0f / lrow[r];
    const int q = qbase + r + lh * 8;
    const size_t rowoff = ((size_t)(bb * SSEQ + q)) * HIDDEN + h * HEADD;
#pragma unroll
    for (int j = 0; j < 8; ++j)
      attn_ws[rowoff + j * 16 + lc] = (__bf16)(o[j][r] * inv);
  }
}

// ---------------------------------------------------------------------------
// Output GEMM: out[4096,2048] = attn(bf16) @ WoutT' + b_out (fp32 out).
// One wave computes a 32x64 tile.
__global__ __launch_bounds__(256) void out_gemm_kernel(const __bf16* __restrict__ A,
                                                       const __bf16* __restrict__ Wt,
                                                       const float* __restrict__ bias,
                                                       float* __restrict__ out) {
  const int lane = threadIdx.x & 31;
  const int wave = threadIdx.x >> 5;
  const int widx = blockIdx.x * 8 + wave;      // 4096 waves total
  const int nstrips = HIDDEN / 64;             // 32
  const int mt = widx / nstrips;               // 0..127
  const int ns = widx % nstrips;
  const int m0 = mt * 32, n0 = ns * 64;

  v8f acc0[4] = {};
  v8f acc1[4] = {};
  for (int kb = 0; kb < HIDDEN; kb += 32) {
    __builtin_prefetch((const void*)(A + (size_t)m0 * HIDDEN + kb + 512), 0, 0);
    v16bf a0 = load_a_frag(A + (size_t)m0 * HIDDEN + kb, HIDDEN, lane);
    v16bf a1 = load_a_frag(A + (size_t)(m0 + 16) * HIDDEN + kb, HIDDEN, lane);
#pragma unroll
    for (int j = 0; j < 4; ++j) {
      v16bf b = load_b_frag(Wt + (size_t)(n0 + j * 16) * HIDDEN + kb, HIDDEN, lane);
      acc0[j] = WMMA_BF16(a0, b, acc0[j]);
      acc1[j] = WMMA_BF16(a1, b, acc1[j]);
    }
  }
  const int lh = lane >> 4, lc = lane & 15;
#pragma unroll
  for (int j = 0; j < 4; ++j) {
    const int n = n0 + j * 16 + lc;
    const float bv = bias[n];
#pragma unroll
    for (int r = 0; r < 8; ++r) {
      const int mA = m0 + r + lh * 8;
      const int mB = m0 + 16 + r + lh * 8;
      out[(size_t)mA * HIDDEN + n] = acc0[j][r] + bv;
      out[(size_t)mB * HIDDEN + n] = acc1[j][r] + bv;
    }
  }
}

// ---------------------------------------------------------------------------
extern "C" void kernel_launch(void* const* d_in, const int* in_sizes, int n_in,
                              void* d_out, int out_size, void* d_ws, size_t ws_size,
                              hipStream_t stream) {
  (void)in_sizes; (void)n_in; (void)out_size; (void)ws_size;
  const float* x     = (const float*)d_in[0];   // [2,2048,2048]
  const float* w_qkv = (const float*)d_in[1];   // [2048,6144]
  const float* b_qkv = (const float*)d_in[2];   // [6144]
  const float* w_out = (const float*)d_in[3];   // [2048,2048]
  const float* b_out = (const float*)d_in[4];   // [2048]
  float* out = (float*)d_out;

  char* ws = (char*)d_ws;
  size_t off = 0;
  __bf16* xb     = (__bf16*)(ws + off); off += (size_t)MROWS * HIDDEN * 2;   // 16 MB
  __bf16* wqkvT  = (__bf16*)(ws + off); off += (size_t)NQKV * HIDDEN * 2;    // 24 MB
  __bf16* woutT  = (__bf16*)(ws + off); off += (size_t)HIDDEN * HIDDEN * 2;  //  8 MB
  __bf16* q_ws   = (__bf16*)(ws + off); off += (size_t)MROWS * HIDDEN * 2;   // 16 MB
  __bf16* k_ws   = (__bf16*)(ws + off); off += (size_t)MROWS * HIDDEN * 2;   // 16 MB
  __bf16* vt_ws  = (__bf16*)(ws + off); off += (size_t)MROWS * HIDDEN * 2;   // 16 MB
  __bf16* att_ws = (__bf16*)(ws + off); off += (size_t)MROWS * HIDDEN * 2;   // 16 MB

  // 1) fp32 -> bf16 conversions (+ tiled weight transposes for contiguous
  //    B-fragment loads)
  {
    long long n = (long long)MROWS * HIDDEN;
    cvt_f32_to_bf16_kernel<<<(unsigned)((n + 255) / 256), 256, 0, stream>>>(x, xb, n);
  }
  {
    const int tiles = (HIDDEN / 32) * (NQKV / 32);    // 12288
    transpose_to_bf16_kernel<<<tiles, 256, 0, stream>>>(w_qkv, wqkvT, HIDDEN, NQKV);
  }
  {
    const int tiles = (HIDDEN / 32) * (HIDDEN / 32);  // 4096
    transpose_to_bf16_kernel<<<tiles, 256, 0, stream>>>(w_out, woutT, HIDDEN, HIDDEN);
  }
  // 2) QKV projection (WMMA, 32x64 wave tiles), scatter to per-head layouts
  {
    const int waves = (MROWS / 32) * (NQKV / 64);   // 12288
    qkv_gemm_kernel<<<waves / 8, 256, 0, stream>>>(xb, wqkvT, b_qkv, q_ws, k_ws, vt_ws);
  }
  // 3) causal flash attention (WMMA)
  {
    const int waves = BBATCH * NHEADS * (SSEQ / 16);  // 4096
    attn_kernel<<<waves / 4, 128, 0, stream>>>(q_ws, k_ws, vt_ws, att_ws);
  }
  // 4) output projection (WMMA, 32x64 wave tiles) -> fp32
  {
    const int waves = (MROWS / 32) * (HIDDEN / 64);   // 4096
    out_gemm_kernel<<<waves / 8, 256, 0, stream>>>(att_ws, woutT, b_out, out);
  }
}
